// GRAPH_AUTOENCODER__70480413327517
// MI455X (gfx1250) — matrix operand order repaired and checked by reading, not compiled, not run on hardware
//
#include <hip/hip_runtime.h>
#include <hip/hip_bf16.h>

// ---------------------------------------------------------------------------
// Graph autoencoder for MI455X (gfx1250).
//  - All 8 dense matmuls use v_wmma_f32_16x16x32_bf16 (bf16 in, fp32 acc).
//  - Edge aggregation uses hardware global_atomic_add_f32 (L2-resident data).
//  - Epilogues (bias / BN*ReLU / residual) fused into GEMM or post kernels,
//    dual fp32+bf16 stores to feed the next WMMA stage without extra passes.
// ---------------------------------------------------------------------------

typedef __bf16 bf16_t;
typedef __attribute__((ext_vector_type(16))) __bf16 v16bf;
typedef __attribute__((ext_vector_type(8)))  __bf16 v8bf;
typedef __attribute__((ext_vector_type(8)))  float  v8f;

#define RN_BN 0.9999950000374997f   // 1/sqrt(1+1e-5), eval-mode BN scale factor

union ABFrag { v16bf v; struct { v8bf lo, hi; } s; };

__device__ __forceinline__ void gatomic_fadd(float* p, float v) {
    unsafeAtomicAdd(p, v);   // lowers to global_atomic_add_f32 on gfx1250
}

// ---------------- small prep kernels ----------------------------------------

__global__ void k_fill(float* p, float val, int n) {
    int i = blockIdx.x * blockDim.x + threadIdx.x;
    if (i < n) p[i] = val;
}

__global__ void k_edge_deg(const int* __restrict__ ei, float* __restrict__ deg, int E) {
    int e = blockIdx.x * blockDim.x + threadIdx.x;
    if (e < E) gatomic_fadd(deg + ei[E + e], 1.0f);   // dst = ei[1][e]
}

__global__ void k_rsqrt_inplace(float* p, int n) {
    int i = blockIdx.x * blockDim.x + threadIdx.x;
    if (i < n) p[i] = rsqrtf(p[i]);
}

__global__ void k_f2b(const float* __restrict__ in, bf16_t* __restrict__ out, long long n) {
    long long i = (long long)blockIdx.x * blockDim.x + threadIdx.x;
    if (i < n) out[i] = (bf16_t)in[i];
}

// W fp32 [K][M] (row-major)  ->  Wt bf16 [M][K]  (so B-fragment K is contiguous)
__global__ void k_prep_w(const float* __restrict__ W, bf16_t* __restrict__ Wt, int K, int M) {
    int i = blockIdx.x * blockDim.x + threadIdx.x;
    if (i >= K * M) return;
    int k = i / M, m = i % M;
    Wt[(long long)m * K + k] = (bf16_t)W[i];
}

// ---------------- WMMA GEMM: [N x Kd] (bf16) @ Wt[Md x Kd] (bf16) -----------
// MODE 0: raw fp32 out ; MODE 1: +bias ; MODE 2: relu(acc+bias)*g*RN + bb (BN)
// outF (fp32) and/or outB (bf16) may be null.
template <int Kd, int Md, int MODE>
__global__ __launch_bounds__(128)
void k_gemm(const bf16_t* __restrict__ A, const bf16_t* __restrict__ Wt,
            const float* __restrict__ bias, const float* __restrict__ g,
            const float* __restrict__ bb,
            float* __restrict__ outF, bf16_t* __restrict__ outB, int N)
{
    constexpr int NT = Md / 16;           // column tiles
    const int wave = threadIdx.x >> 5;
    const int lane = threadIdx.x & 31;
    const int m0 = (blockIdx.x * 4 + wave) * 16;
    if (m0 >= N) return;                  // uniform per wave: EXEC stays all-ones
    const int half = lane >> 4;           // 0: lanes 0-15, 1: lanes 16-31
    const int l16  = lane & 15;

    v8f acc[NT] = {};

    // A fragment: 16-bit 16x32 layout => per lane two contiguous 16B loads
    int arow = m0 + l16; if (arow >= N) arow = N - 1;
    const bf16_t* ap = A + (long long)arow * Kd;

    for (int kc = 0; kc < Kd; kc += 32) {
        ABFrag af;
        af.s.lo = *(const v8bf*)(ap + kc + half * 16);       // K = kc+{0..7 | 16..23}
        af.s.hi = *(const v8bf*)(ap + kc + half * 16 + 8);   // K = kc+{8..15 | 24..31}
        v16bf a = af.v;
#pragma unroll
        for (int t = 0; t < NT; ++t) {
            // B fragment: lane n = l16, 16 contiguous K values from transposed W
            const bf16_t* bp = Wt + (long long)(t * 16 + l16) * Kd + kc + half * 16;
            v16bf b = *(const v16bf*)bp;
            acc[t] = __builtin_amdgcn_wmma_f32_16x16x32_bf16(
                false, a, false, b, (short)0, acc[t], false, false);
        }
    }

    // Epilogue: C/D layout -> lane holds rows m0+half*8+r, col = tile*16+l16
    const int rbase = m0 + half * 8;
#pragma unroll
    for (int t = 0; t < NT; ++t) {
        const int col = t * 16 + l16;
        const float bi = (MODE >= 1) ? bias[col] : 0.0f;
        const float sc = (MODE == 2) ? g[col] * RN_BN : 1.0f;
        const float sh = (MODE == 2) ? bb[col] : 0.0f;
#pragma unroll
        for (int r = 0; r < 8; ++r) {
            const int row = rbase + r;
            if (row >= N) continue;
            float v = acc[t][r] + bi;
            if (MODE == 2) v = fmaxf(v, 0.0f) * sc + sh;
            const long long o = (long long)row * Md + col;
            if (outF) outF[o] = v;
            if (outB) outB[o] = (bf16_t)v;
        }
    }
}

// ---------------- GCN aggregation -------------------------------------------

// agg[n][f] = H[n][f] * dinv[n]^2 + b[f]     (self-loop term + bias, seeds atomics)
__global__ void k_self_loop(const float* __restrict__ H, const float* __restrict__ dinv,
                            const float* __restrict__ bias, float* __restrict__ agg,
                            int N, int F)
{
    long long i = (long long)blockIdx.x * blockDim.x + threadIdx.x;
    if (i >= (long long)N * F) return;
    int n = (int)(i / F), f = (int)(i % F);
    float di = dinv[n];
    agg[i] = H[i] * di * di + bias[f];
}

// agg[dst] += H[src] * dinv[src]*dinv[dst]; one lane = 4 feats, one wave ~ 1 edge
__global__ void k_edge_agg(const float* __restrict__ H, const int* __restrict__ ei,
                           const float* __restrict__ dinv, float* __restrict__ agg,
                           int E, int F)
{
    const int CH = F >> 2;                     // float4 chunks per edge
    long long tid = (long long)blockIdx.x * blockDim.x + threadIdx.x;
    int e = (int)(tid / CH);
    int c = (int)(tid % CH);
    if (e >= E) return;
    const int s = ei[e];                       // src row
    const int d = ei[E + e];                   // dst row
    const float coef = dinv[s] * dinv[d];
    const float4 hv = *(const float4*)(H + (long long)s * F + c * 4);
    float* apx = agg + (long long)d * F + c * 4;
    gatomic_fadd(apx + 0, hv.x * coef);
    gatomic_fadd(apx + 1, hv.y * coef);
    gatomic_fadd(apx + 2, hv.z * coef);
    gatomic_fadd(apx + 3, hv.w * coef);
}

// h = relu( relu(bn(agg)) + residual ); dual fp32 + optional bf16 store
__global__ void k_post(const float* __restrict__ agg, const float* __restrict__ res,
                       const float* __restrict__ g, const float* __restrict__ bb,
                       float* __restrict__ hF, bf16_t* __restrict__ hB, int N, int F)
{
    long long i = (long long)blockIdx.x * blockDim.x + threadIdx.x;
    if (i >= (long long)N * F) return;
    int f = (int)(i % F);
    float c = fmaxf(agg[i] * (g[f] * RN_BN) + bb[f], 0.0f);
    float h = fmaxf(c + res[i], 0.0f);
    hF[i] = h;
    if (hB) hB[i] = (bf16_t)h;
}

// z = h / max(||h||, 1e-12), F = 64, one wave per node, write bf16 for decoder
__global__ void k_l2norm(const float* __restrict__ h, bf16_t* __restrict__ zB, int N)
{
    int node = blockIdx.x * 8 + (threadIdx.x >> 5);
    int lane = threadIdx.x & 31;
    if (node >= N) return;
    float2 v = *(const float2*)(h + (long long)node * 64 + lane * 2);
    float ss = v.x * v.x + v.y * v.y;
#pragma unroll
    for (int off = 16; off > 0; off >>= 1) ss += __shfl_xor(ss, off, 32);
    float sc = 1.0f / fmaxf(sqrtf(ss), 1e-12f);
    bf16_t* o = zB + (long long)node * 64 + lane * 2;
    o[0] = (bf16_t)(v.x * sc);
    o[1] = (bf16_t)(v.y * sc);
}

// ---------------------------------------------------------------------------

static inline int cdiv(long long a, long long b) { return (int)((a + b - 1) / b); }

extern "C" void kernel_launch(void* const* d_in, const int* in_sizes, int n_in,
                              void* d_out, int out_size, void* d_ws, size_t ws_size,
                              hipStream_t stream)
{
    const float* x  = (const float*)d_in[0];
    const int*   ei = (const int*)d_in[1];                 // edge_index [2][E]
    const float* enc_W[3]  = {(const float*)d_in[2],  (const float*)d_in[6],  (const float*)d_in[10]};
    const float* enc_b[3]  = {(const float*)d_in[3],  (const float*)d_in[7],  (const float*)d_in[11]};
    const float* enc_g[3]  = {(const float*)d_in[4],  (const float*)d_in[8],  (const float*)d_in[12]};
    const float* enc_bb[3] = {(const float*)d_in[5],  (const float*)d_in[9],  (const float*)d_in[13]};
    const float* sc_W = (const float*)d_in[14];
    const float* sc_b = (const float*)d_in[15];
    const float* dec_W[4]  = {(const float*)d_in[16], (const float*)d_in[20], (const float*)d_in[24], (const float*)d_in[28]};
    const float* dec_b[4]  = {(const float*)d_in[17], (const float*)d_in[21], (const float*)d_in[25], (const float*)d_in[29]};
    const float* dec_g[3]  = {(const float*)d_in[18], (const float*)d_in[22], (const float*)d_in[26]};
    const float* dec_bb[3] = {(const float*)d_in[19], (const float*)d_in[23], (const float*)d_in[27]};

    const int N = in_sizes[0] / 128;      // 50000
    const int E = in_sizes[1] / 2;        // 800000

    // ---- workspace carve (256B aligned) ----
    size_t off = 0;
    char* base = (char*)d_ws;
    auto carve = [&](size_t bytes) -> void* {
        void* p = base + off;
        off = (off + bytes + 255) & ~(size_t)255;
        return p;
    };
    float*  F0  = (float*)carve((size_t)N * 128 * 4);   // act ping / GEMM out
    float*  F1  = (float*)carve((size_t)N * 128 * 4);   // act pong / GEMM out
    float*  F2  = (float*)carve((size_t)N * 128 * 4);   // agg (encoder) / B1 (decoder)
    float*  SC  = (float*)carve((size_t)N * 64 * 4);    // block-2 linear shortcut
    bf16_t* B0  = (bf16_t*)carve((size_t)N * 128 * 2);  // bf16 activation staging
    float*  DEG = (float*)carve((size_t)N * 4);         // degree -> dinv (in place)
    bf16_t* WTe0 = (bf16_t*)carve(128 * 128 * 2);
    bf16_t* WTe1 = (bf16_t*)carve(128 * 128 * 2);
    bf16_t* WTe2 = (bf16_t*)carve(128 * 64 * 2);
    bf16_t* WTsc = (bf16_t*)carve(128 * 64 * 2);
    bf16_t* WTd0 = (bf16_t*)carve(64 * 64 * 2);
    bf16_t* WTd1 = (bf16_t*)carve(64 * 128 * 2);
    bf16_t* WTd2 = (bf16_t*)carve(128 * 128 * 2);
    bf16_t* WTd3 = (bf16_t*)carve(128 * 128 * 2);
    bf16_t* B1  = (bf16_t*)F2;                          // alias: agg dead in decoder

    // ---- weight convert + transpose (tiny) ----
    k_prep_w<<<cdiv(128 * 128, 256), 256, 0, stream>>>(enc_W[0], WTe0, 128, 128);
    k_prep_w<<<cdiv(128 * 128, 256), 256, 0, stream>>>(enc_W[1], WTe1, 128, 128);
    k_prep_w<<<cdiv(128 * 64, 256), 256, 0, stream>>>(enc_W[2], WTe2, 128, 64);
    k_prep_w<<<cdiv(128 * 64, 256), 256, 0, stream>>>(sc_W,     WTsc, 128, 64);
    k_prep_w<<<cdiv(64 * 64, 256), 256, 0, stream>>>(dec_W[0],  WTd0, 64, 64);
    k_prep_w<<<cdiv(64 * 128, 256), 256, 0, stream>>>(dec_W[1], WTd1, 64, 128);
    k_prep_w<<<cdiv(128 * 128, 256), 256, 0, stream>>>(dec_W[2], WTd2, 128, 128);
    k_prep_w<<<cdiv(128 * 128, 256), 256, 0, stream>>>(dec_W[3], WTd3, 128, 128);

    // ---- symmetric-norm degree: deg = 1 + indeg; dinv = rsqrt(deg) ----
    k_fill<<<cdiv(N, 256), 256, 0, stream>>>(DEG, 1.0f, N);
    k_edge_deg<<<cdiv(E, 256), 256, 0, stream>>>(ei, DEG, E);
    k_rsqrt_inplace<<<cdiv(N, 256), 256, 0, stream>>>(DEG, N);

    // ---- x -> bf16 ----
    const long long NF = (long long)N * 128;
    k_f2b<<<cdiv(NF, 256), 256, 0, stream>>>(x, B0, NF);

    const int GB = cdiv(N, 64);   // GEMM: 4 waves/block, 16 rows/wave

    // ---- encoder block 0 (128 -> 128, residual = x) ----
    k_gemm<128, 128, 0><<<GB, 128, 0, stream>>>(B0, WTe0, nullptr, nullptr, nullptr, F0, nullptr, N);
    k_self_loop<<<cdiv(NF, 256), 256, 0, stream>>>(F0, DEG, enc_b[0], F2, N, 128);
    k_edge_agg<<<cdiv((long long)E * 32, 256), 256, 0, stream>>>(F0, ei, DEG, F2, E, 128);
    k_post<<<cdiv(NF, 256), 256, 0, stream>>>(F2, x, enc_g[0], enc_bb[0], F0, B0, N, 128);

    // ---- encoder block 1 (128 -> 128, residual = h0 in F0) ----
    k_gemm<128, 128, 0><<<GB, 128, 0, stream>>>(B0, WTe1, nullptr, nullptr, nullptr, F1, nullptr, N);
    k_self_loop<<<cdiv(NF, 256), 256, 0, stream>>>(F1, DEG, enc_b[1], F2, N, 128);
    k_edge_agg<<<cdiv((long long)E * 32, 256), 256, 0, stream>>>(F1, ei, DEG, F2, E, 128);
    k_post<<<cdiv(NF, 256), 256, 0, stream>>>(F2, F0, enc_g[1], enc_bb[1], F1, B0, N, 128);

    // ---- encoder block 2 (128 -> 64, residual = h1 @ sc_W + sc_b) ----
    const long long NF64 = (long long)N * 64;
    k_gemm<128, 64, 0><<<GB, 128, 0, stream>>>(B0, WTe2, nullptr, nullptr, nullptr, F0, nullptr, N);
    k_gemm<128, 64, 1><<<GB, 128, 0, stream>>>(B0, WTsc, sc_b, nullptr, nullptr, SC, nullptr, N);
    k_self_loop<<<cdiv(NF64, 256), 256, 0, stream>>>(F0, DEG, enc_b[2], F2, N, 64);
    k_edge_agg<<<cdiv((long long)E * 16, 256), 256, 0, stream>>>(F0, ei, DEG, F2, E, 64);
    k_post<<<cdiv(NF64, 256), 256, 0, stream>>>(F2, SC, enc_g[2], enc_bb[2], F0, nullptr, N, 64);

    // ---- L2 row-normalize -> z (bf16 in B0) ----
    k_l2norm<<<cdiv(N, 8), 256, 0, stream>>>(F0, B0, N);

    // ---- decoder: (Linear+ReLU+BN) x3, final Linear -> d_out fp32 ----
    k_gemm<64, 64, 2><<<GB, 128, 0, stream>>>(B0, WTd0, dec_b[0], dec_g[0], dec_bb[0], nullptr, B1, N);
    k_gemm<64, 128, 2><<<GB, 128, 0, stream>>>(B1, WTd1, dec_b[1], dec_g[1], dec_bb[1], nullptr, B0, N);
    k_gemm<128, 128, 2><<<GB, 128, 0, stream>>>(B0, WTd2, dec_b[2], dec_g[2], dec_bb[2], nullptr, B1, N);
    k_gemm<128, 128, 1><<<GB, 128, 0, stream>>>(B1, WTd3, dec_b[3], nullptr, nullptr, (float*)d_out, nullptr, N);
}
